// GINModel_59390807769652
// MI455X (gfx1250) — matrix operand order repaired and checked by reading, not compiled, not run on hardware
//
#include <hip/hip_runtime.h>

#define NN 100000
#define NE 1600000
#define NG 64
#define DIM 128
#define NL 4
#define OUTD 32

typedef __bf16 bf16_t;
typedef __attribute__((ext_vector_type(8)))  bf16_t v8bf;
typedef __attribute__((ext_vector_type(16))) bf16_t v16bf;
typedef __attribute__((ext_vector_type(8)))  float  v8f;

__device__ __forceinline__ unsigned short f32_to_bf16_rne(float f) {
    union { float f; unsigned u; } x; x.f = f;
    unsigned r = x.u + 0x7FFFu + ((x.u >> 16) & 1u);
    return (unsigned short)(r >> 16);
}

// ---------------- utility kernels ----------------
__global__ void zero_f32_kernel(float* __restrict__ p, int n) {
    int i = blockIdx.x * blockDim.x + threadIdx.x;
    if (i < n) p[i] = 0.0f;
}

__global__ void deg_kernel(const int* __restrict__ dst, float* __restrict__ deg, int nE) {
    int e = blockIdx.x * blockDim.x + threadIdx.x;
    if (e < nE) atomicAdd(deg + dst[e], 1.0f);
}

// transpose + convert weights: W[l][k][n] (f32) -> Wt[l][n][k] (bf16)
__global__ void wprep_kernel(const float* __restrict__ W, unsigned short* __restrict__ Wt, int total) {
    int i = blockIdx.x * blockDim.x + threadIdx.x;
    if (i >= total) return;
    int l = i >> 14;
    int rem = i & 16383;
    int k = rem >> 7;
    int n = rem & 127;
    Wt[(l << 14) + (n << 7) + k] = f32_to_bf16_rne(W[i]);
}

// scatter-add h[src] into agg[dst]; one edge per 32 lanes, 4 floats per lane
__global__ void edge_agg_kernel(const float* __restrict__ h, const int* __restrict__ src,
                                const int* __restrict__ dst, float* __restrict__ agg, int nE) {
    int t = blockIdx.x * blockDim.x + threadIdx.x;
    int e = t >> 5;
    int lane = t & 31;
    if (e >= nE) return;
    int so = src[e] * DIM + lane * 4;
    int dofs = dst[e] * DIM + lane * 4;
    float4 v = *(const float4*)(h + so);
    atomicAdd(agg + dofs + 0, v.x);
    atomicAdd(agg + dofs + 1, v.y);
    atomicAdd(agg + dofs + 2, v.z);
    atomicAdd(agg + dofs + 3, v.w);
}

// z = h + agg/deg  -> bf16
__global__ void zprep_kernel(const float* __restrict__ hin, const float* __restrict__ agg,
                             const float* __restrict__ deg, unsigned short* __restrict__ zb, int total) {
    int i = blockIdx.x * blockDim.x + threadIdx.x;
    if (i >= total) return;
    int node = i >> 7;
    float dg = deg[node];
    dg = dg < 1.0f ? 1.0f : dg;
    zb[i] = f32_to_bf16_rne(hin[i] + agg[i] / dg);
}

// ---------------- WMMA GEMM: [N,128](bf16) x Wt[n][k](bf16) + bias ----------------
// one wave -> 16 rows x 128 cols (8 column tiles), K=128 in 4 steps of 32
template <bool RELU_BF16_OUT>
__global__ __launch_bounds__(256) void gemm128_kernel(const unsigned short* __restrict__ Xus,
                                                      const unsigned short* __restrict__ Wtus,
                                                      const float* __restrict__ bias,
                                                      void* __restrict__ Y, int nRowTiles) {
    int wave = blockIdx.x * (blockDim.x >> 5) + (threadIdx.x >> 5);
    if (wave >= nRowTiles) return;   // uniform per wave: EXEC stays all-ones
    const bf16_t* X  = (const bf16_t*)Xus;
    const bf16_t* Wt = (const bf16_t*)Wtus;
    int lane = threadIdx.x & 31;
    int m    = lane & 15;
    int half = lane >> 4;            // 0: lanes 0-15, 1: lanes 16-31
    int base = wave * 16;
    int arow = base + m;

    v8f zero8 = {0.f, 0.f, 0.f, 0.f, 0.f, 0.f, 0.f, 0.f};
    v8f acc[8];
#pragma unroll
    for (int ct = 0; ct < 8; ++ct) acc[ct] = zero8;

#pragma unroll
    for (int kb = 0; kb < DIM; kb += 32) {
        // A fragment (16x32 bf16): lane holds K = c..c+15 contiguous, c = kb (+16 for upper half)
        int ka = kb + half * 16;
        v8bf alo = *(const v8bf*)(X + arow * DIM + ka);
        v8bf ahi = *(const v8bf*)(X + arow * DIM + ka + 8);
        v16bf afrag = __builtin_shufflevector(alo, ahi, 0, 1, 2, 3, 4, 5, 6, 7,
                                              8, 9, 10, 11, 12, 13, 14, 15);
#pragma unroll
        for (int ct = 0; ct < 8; ++ct) {
            // B fragment (32x16 bf16): lane holds col (lane&15), K = kb + 16*half .. +15 (contiguous in Wt)
            int ncol = ct * 16 + m;
            v16bf bfrag = *(const v16bf*)(Wt + ncol * DIM + kb + half * 16);
            acc[ct] = __builtin_amdgcn_wmma_f32_16x16x32_bf16(
                false, afrag, false, bfrag, (short)0, acc[ct], false, false);
        }
    }

    // C layout: VGPR r, lanes 0-15 -> (M=r, N=lane); lanes 16-31 -> (M=8+r, N=lane-16)
#pragma unroll
    for (int ct = 0; ct < 8; ++ct) {
        int ocol = ct * 16 + m;
        float b = bias[ocol];
#pragma unroll
        for (int r = 0; r < 8; ++r) {
            int orow = base + r + half * 8;
            float v = acc[ct][r] + b;
            if (RELU_BF16_OUT) {
                v = v > 0.f ? v : 0.f;
                ((unsigned short*)Y)[orow * DIM + ocol] = f32_to_bf16_rne(v);
            } else {
                ((float*)Y)[orow * DIM + ocol] = v;
            }
        }
    }
}

// ---------------- BatchNorm ----------------
__global__ void bn_stats_kernel(const float* __restrict__ z, float* __restrict__ bsum,
                                float* __restrict__ bsq, int n) {
    int c = threadIdx.x; // 128 threads
    int rowsPerBlock = (n + gridDim.x - 1) / gridDim.x;
    int r0 = blockIdx.x * rowsPerBlock;
    int r1 = r0 + rowsPerBlock;
    if (r1 > n) r1 = n;
    float s = 0.f, s2 = 0.f;
    for (int r = r0; r < r1; ++r) {
        float v = z[r * DIM + c];
        s += v;
        s2 += v * v;
    }
    atomicAdd(bsum + c, s);
    atomicAdd(bsq + c, s2);
}

__global__ void bn_apply_kernel(const float* __restrict__ z, const float* __restrict__ bsum,
                                const float* __restrict__ bsq, const float* __restrict__ gamma,
                                const float* __restrict__ beta, float* __restrict__ hout, int total) {
    int i = blockIdx.x * blockDim.x + threadIdx.x;
    if (i >= total) return;
    int c = i & 127;
    const float invN = 1.0f / (float)NN;
    float mu = bsum[c] * invN;
    float var = bsq[c] * invN - mu * mu;
    float v = gamma[c] * (z[i] - mu) * rsqrtf(var + 1e-5f) + beta[c];
    hout[i] = v > 0.f ? v : 0.f;
}

// ---------------- readout + projection ----------------
__global__ void readout_kernel(const float* __restrict__ h, const int* __restrict__ gid,
                               float* __restrict__ hg, float* __restrict__ cnt) {
    int node = blockIdx.x;
    int c = threadIdx.x; // 128
    int g = gid[node];
    atomicAdd(hg + g * DIM + c, h[node * DIM + c]);
    if (c == 0) atomicAdd(cnt + g, 1.0f);
}

__global__ void proj_kernel(const float* __restrict__ hg, const float* __restrict__ cnt,
                            const float* __restrict__ Wp, const float* __restrict__ bp,
                            float* __restrict__ out) {
    int g = blockIdx.x;  // 64
    int o = threadIdx.x; // 32
    float cn = cnt[g];
    cn = cn < 1.0f ? 1.0f : cn;
    float acc = bp[o];
    for (int d = 0; d < DIM; ++d) acc += (hg[g * DIM + d] / cn) * Wp[d * OUTD + o];
    out[g * OUTD + o] = acc;
}

// ---------------- host launch ----------------
extern "C" void kernel_launch(void* const* d_in, const int* in_sizes, int n_in,
                              void* d_out, int out_size, void* d_ws, size_t ws_size,
                              hipStream_t stream) {
    const float* features = (const float*)d_in[0];
    const float* W1    = (const float*)d_in[1];
    const float* b1    = (const float*)d_in[2];
    const float* W2    = (const float*)d_in[3];
    const float* b2    = (const float*)d_in[4];
    const float* gamma = (const float*)d_in[5];
    const float* beta  = (const float*)d_in[6];
    const float* Wp    = (const float*)d_in[7];
    const float* bp    = (const float*)d_in[8];
    const int* src = (const int*)d_in[9];
    const int* dst = (const int*)d_in[10];
    const int* gid = (const int*)d_in[11];
    float* out = (float*)d_out;

    // workspace carve-out (256B aligned)
    size_t off = 0;
    char* base = (char*)d_ws;
    auto carve = [&](size_t bytes) -> void* {
        void* p = base + off;
        off += (bytes + 255) & ~(size_t)255;
        return p;
    };
    float* h            = (float*)carve((size_t)NN * DIM * 4);
    float* aggz         = (float*)carve((size_t)NN * DIM * 4); // agg, then GEMM2 output
    unsigned short* zb  = (unsigned short*)carve((size_t)NN * DIM * 2);
    unsigned short* tb  = (unsigned short*)carve((size_t)NN * DIM * 2);
    float* deg          = (float*)carve((size_t)NN * 4);
    unsigned short* w1b = (unsigned short*)carve((size_t)NL * DIM * DIM * 2);
    unsigned short* w2b = (unsigned short*)carve((size_t)NL * DIM * DIM * 2);
    float* bsum         = (float*)carve(DIM * 4);
    float* bsq          = (float*)carve(DIM * 4);
    float* hg           = (float*)carve((size_t)NG * DIM * 4);
    float* cnt          = (float*)carve((size_t)NG * 4);

    const int ND = NN * DIM;           // 12.8M
    const int nRowTiles = NN / 16;     // 6250
    const int gemmBlocks = (nRowTiles + 7) / 8;

    // degrees
    zero_f32_kernel<<<(NN + 255) / 256, 256, 0, stream>>>(deg, NN);
    deg_kernel<<<(NE + 255) / 256, 256, 0, stream>>>(dst, deg, NE);

    // weight prep (bf16 + transpose)
    wprep_kernel<<<(NL * DIM * DIM + 255) / 256, 256, 0, stream>>>(W1, w1b, NL * DIM * DIM);
    wprep_kernel<<<(NL * DIM * DIM + 255) / 256, 256, 0, stream>>>(W2, w2b, NL * DIM * DIM);

    for (int l = 0; l < NL; ++l) {
        const float* hin = (l == 0) ? features : h;

        zero_f32_kernel<<<(ND + 255) / 256, 256, 0, stream>>>(aggz, ND);
        edge_agg_kernel<<<(NE * 32) / 256, 256, 0, stream>>>(hin, src, dst, aggz, NE);
        zprep_kernel<<<(ND + 255) / 256, 256, 0, stream>>>(hin, aggz, deg, zb, ND);

        gemm128_kernel<true><<<gemmBlocks, 256, 0, stream>>>(
            zb, w1b + (size_t)l * DIM * DIM, b1 + l * DIM, (void*)tb, nRowTiles);
        gemm128_kernel<false><<<gemmBlocks, 256, 0, stream>>>(
            tb, w2b + (size_t)l * DIM * DIM, b2 + l * DIM, (void*)aggz, nRowTiles);

        zero_f32_kernel<<<1, DIM, 0, stream>>>(bsum, DIM);
        zero_f32_kernel<<<1, DIM, 0, stream>>>(bsq, DIM);
        bn_stats_kernel<<<256, DIM, 0, stream>>>(aggz, bsum, bsq, NN);
        bn_apply_kernel<<<(ND + 255) / 256, 256, 0, stream>>>(
            aggz, bsum, bsq, gamma + l * DIM, beta + l * DIM, h, ND);
    }

    zero_f32_kernel<<<(NG * DIM + 255) / 256, 256, 0, stream>>>(hg, NG * DIM);
    zero_f32_kernel<<<1, NG, 0, stream>>>(cnt, NG);
    readout_kernel<<<NN, DIM, 0, stream>>>(h, gid, hg, cnt);
    proj_kernel<<<NG, OUTD, 0, stream>>>(hg, cnt, Wp, bp, out);
}
